// SimpleNetGRU_76484777607659
// MI455X (gfx1250) — matrix operand order repaired and checked by reading, not compiled, not run on hardware
//
#include <hip/hip_runtime.h>
#include <hip/hip_bf16.h>

// ---------------- model dims ----------------
#define Bsz   64
#define Tt    512
#define Dd    64
#define Ff    64
#define Uu    128
#define NHh   4
#define KDd   128
#define Mm    256          // 2U
#define G3x   384          // 3U
#define QKVd  512          // NH*KD
#define BT    (Bsz * Tt)   // 32768

typedef __attribute__((ext_vector_type(16))) _Float16 v16h;
typedef __attribute__((ext_vector_type(8)))  _Float16 v8h;
typedef __attribute__((ext_vector_type(4)))  _Float16 v4h;
typedef __attribute__((ext_vector_type(8)))  float    v8f;

// ---- async global->LDS copies (gfx1250, ASYNCcnt) with sync fallback ----
// Probe-learned signature: (int4*, int4*, imm offset, imm cpol); generic ptrs.
#if __has_builtin(__builtin_amdgcn_global_load_async_to_lds_b128) && \
    __has_builtin(__builtin_amdgcn_s_wait_asynccnt)
typedef int async_b128_t __attribute__((vector_size(16)));
#define ASYNC_COPY_B128(gsrc, ldst)                                  \
  __builtin_amdgcn_global_load_async_to_lds_b128(                    \
      (async_b128_t*)(gsrc), (async_b128_t*)(ldst), 0, 0)
#define ASYNC_WAIT() __builtin_amdgcn_s_wait_asynccnt(0)
#else
#define ASYNC_COPY_B128(gsrc, ldst)                  \
  do {                                               \
    *(v8h*)(ldst) = *(const v8h*)(gsrc);             \
  } while (0)
#define ASYNC_WAIT() \
  do {               \
  } while (0)
#endif

__device__ __forceinline__ v8f wmma_f16(v16h a, v16h b, v8f c) {
  return __builtin_amdgcn_wmma_f32_16x16x32_f16(false, a, false, b, (short)0, c,
                                                false, false);
}

// Fragment slot maps (16x16x32 f16 WMMA, wave32):
//  A (16x32): lane = m + 16*h, h=(k>>3)&1 ; slot = (k&7) + ((k&16)>>1)
//  B (32x16): lane = n + 16*((kk>>3)&1) ; slot = (kk&7) + ((kk&16)>>1)   (kk = k mod 32)
// One 32-byte record per lane -> single v16h load per fragment.

// ---------------- pack W[K][N] f32 -> B-fragment-layout f16 ----------------
__global__ __launch_bounds__(256) void pack_b_kernel(const float* __restrict__ W,
                                                     _Float16* __restrict__ out,
                                                     int K, int N) {
  const int idx = blockIdx.x * blockDim.x + threadIdx.x;
  if (idx >= K * N) return;
  const int k = idx / N, n = idx % N;
  const int kb = k >> 5, kk = k & 31, nb = n >> 4;
  const int lane = (n & 15) + (((kk >> 3) & 1) << 4);
  const int slot = (kk & 7) + ((kk & 16) >> 1);
  out[(((size_t)kb * (N >> 4) + nb) << 9) + (lane << 4) + slot] =
      (_Float16)W[idx];
}

// ---------------- input f32 -> f16 with 1-row zero pad on both ends --------
__global__ __launch_bounds__(256) void cvt_pad_kernel(const float* __restrict__ in,
                                                      _Float16* __restrict__ out) {
  const int idx = blockIdx.x * blockDim.x + threadIdx.x;
  const int total = Bsz * (Tt + 2) * Dd;
  if (idx >= total) return;
  const int d = idx % Dd;
  const int p = (idx / Dd) % (Tt + 2);
  const int b = idx / (Dd * (Tt + 2));
  const float v =
      (p == 0 || p == Tt + 1) ? 0.0f : in[((size_t)b * Tt + (p - 1)) * Dd + d];
  out[idx] = (_Float16)v;
}

// ---------------- fold conv bias + BN into per-channel scale/shift ---------
__global__ void prep_bn_kernel(const float* __restrict__ cb,
                               const float* __restrict__ g,
                               const float* __restrict__ be,
                               const float* __restrict__ mu,
                               const float* __restrict__ var,
                               float* __restrict__ sc, float* __restrict__ sh) {
  const int f = blockIdx.x * blockDim.x + threadIdx.x;
  if (f >= Ff) return;
  const float s = g[f] * rsqrtf(var[f] + 1e-3f);
  sc[f] = s;
  sh[f] = be[f] + (cb[f] - mu[f]) * s;
}

// ---------------- generic WMMA GEMM ----------------
// out = epilogue(X * Wp); X row-major f16 (row stride lda, per-batch rowpad),
// Wp pre-packed B-fragment layout. NTW = n-tiles per wave (BN = 32*NTW).
#define BMT 64
#define BKT 32
template <int NTW>
__global__ __launch_bounds__(256) void gemm_wmma_kernel(
    const _Float16* __restrict__ X, const _Float16* __restrict__ Wp,
    const float* __restrict__ bias, const float* __restrict__ resid,
    const float* __restrict__ scale, const float* __restrict__ shift,
    float* __restrict__ outF, _Float16* __restrict__ outH,
    _Float16* __restrict__ outT,  // per-head transposed f16 [b,h,kd,t] (V path)
    int Mrows, int K, int N, int lda, int rowpad) {
  __shared__ __align__(32) _Float16 Af[4][32][16];
  const int tid = threadIdx.x, wave = tid >> 5, l = tid & 31;
  const int m0 = blockIdx.y * BMT, n0 = blockIdx.x * (32 * NTW);
  const int mt = wave & 3;
  const int ntb = (wave >> 2) * NTW;
  const int nblk = N >> 4;
  v8f acc[NTW] = {};

  const int arow = tid >> 2, acg = (tid & 3) << 3;
  const int amt = arow >> 4;
  const int alane = (arow & 15) + (((acg >> 3) & 1) << 4);
  const int asb = (acg & 16) >> 1;
  const int xr = m0 + arow;
  const _Float16* xbase = X + (size_t)xr * lda + (size_t)(xr / Tt) * rowpad;

  for (int k0 = 0; k0 < K; k0 += BKT) {
    ASYNC_COPY_B128(&xbase[k0 + acg], &Af[amt][alane][asb]);
    if (k0 + BKT < K) __builtin_prefetch(&xbase[k0 + BKT + acg], 0, 0);
    ASYNC_WAIT();
    __syncthreads();
    v16h a = *(const v16h*)&Af[mt][l][0];
    const _Float16* wb =
        &Wp[(((size_t)(k0 >> 5) * nblk + (n0 >> 4) + ntb) << 9) + (l << 4)];
#pragma unroll
    for (int j = 0; j < NTW; ++j) {
      v16h b = *(const v16h*)&wb[(size_t)j << 9];
      acc[j] = wmma_f16(a, b, acc[j]);
    }
    __syncthreads();
  }
  const int n = l & 15, mh = l >> 4;
#pragma unroll
  for (int j = 0; j < NTW; ++j) {
    const int ncol = n0 + (ntb + j) * 16 + n;
    const float bv = bias ? bias[ncol] : 0.0f;
    const float sc = scale ? scale[ncol] : 1.0f;
    const float sf = shift ? shift[ncol] : 0.0f;
#pragma unroll
    for (int i = 0; i < 8; ++i) {
      const int row = m0 + mt * 16 + i + 8 * mh;
      float v = acc[j][i] + bv;
      if (resid) v += resid[(size_t)row * N + ncol];
      if (scale) v = fmaxf(v * sc + sf, 0.0f);  // fused BN + ReLU
      if (outF) outF[(size_t)row * N + ncol] = v;
      if (outH) outH[(size_t)row * N + ncol] = (_Float16)v;
      if (outT) {
        const int bb = row / Tt, t = row % Tt;
        const int hh = ncol / KDd, kd = ncol % KDd;
        outT[(((size_t)bb * NHh + hh) * KDd + kd) * Tt + t] = (_Float16)v;
      }
    }
  }
}

// ---------------- GRU recurrence (16 batch rows / WG) ----------------
__global__ __launch_bounds__(256) void gru_seq_kernel(
    const float* __restrict__ xg,      // [B*T, 3U]
    const _Float16* __restrict__ Whp,  // packed [U x 3U]
    const float* __restrict__ bh,      // [3U]
    float* __restrict__ seqF, _Float16* __restrict__ seqH, int seq_stride,
    int seq_coloff, float* __restrict__ hfin, int h_stride, int h_coloff,
    int reverse) {
  __shared__ __align__(32) _Float16 hHf[4][32][16];  // h in A-frag layout
  __shared__ float hF[16][Uu];
  __shared__ float hg[16][G3x + 1];
  const int tid = threadIdx.x, wave = tid >> 5, l = tid & 31;
  const int b0 = blockIdx.x * 16;
  for (int i = tid; i < 4 * 32 * 16; i += 256) ((_Float16*)hHf)[i] = (_Float16)0.0f;
  for (int i = tid; i < 16 * Uu; i += 256) hF[i / Uu][i % Uu] = 0.0f;
  __syncthreads();

  const int n = l & 15, mh = l >> 4;
  const int r = tid >> 4, cb = (tid & 15) << 3;
  const int gkb = cb >> 5;
  const int glane = r + (((cb >> 3) & 1) << 4);
  const int gsb = (cb & 16) >> 1;

  for (int step = 0; step < Tt; ++step) {
    const int t = reverse ? (Tt - 1 - step) : step;
    v8f acc[3] = {};
#pragma unroll
    for (int ks = 0; ks < 4; ++ks) {
      v16h a = *(const v16h*)&hHf[ks][l][0];
#pragma unroll
      for (int j = 0; j < 3; ++j) {
        const int nt = wave * 3 + j;
        v16h b = *(const v16h*)&Whp[(((size_t)ks * 24 + nt) << 9) + (l << 4)];
        acc[j] = wmma_f16(a, b, acc[j]);
      }
    }
#pragma unroll
    for (int j = 0; j < 3; ++j) {
      const int nc = (wave * 3 + j) * 16 + n;
      const float bv = bh[nc];
#pragma unroll
      for (int i = 0; i < 8; ++i) hg[i + 8 * mh][nc] = acc[j][i] + bv;
    }
    __syncthreads();
    const size_t xrow = ((size_t)(b0 + r) * Tt + t) * G3x;
    const v8f xzv = *(const v8f*)&xg[xrow + cb];
    const v8f xrv = *(const v8f*)&xg[xrow + Uu + cb];
    const v8f xnv = *(const v8f*)&xg[xrow + 2 * Uu + cb];
    v8f hvv;
    v8h hhv;
#pragma unroll
    for (int q = 0; q < 8; ++q) {
      const int c = cb + q;
      const float hz = hg[r][c], hr = hg[r][Uu + c], hn = hg[r][2 * Uu + c];
      const float z  = 1.0f / (1.0f + __expf(-(xzv[q] + hz)));
      const float rg = 1.0f / (1.0f + __expf(-(xrv[q] + hr)));
      const float nn = tanhf(xnv[q] + rg * hn);
      const float hv = z * hF[r][c] + (1.0f - z) * nn;
      hvv[q] = hv;
      hhv[q] = (_Float16)hv;
      hF[r][c] = hv;
    }
    *(v8h*)&hHf[gkb][glane][gsb] = hhv;
    if (seqF)
      *(v8f*)&seqF[((size_t)(b0 + r) * Tt + t) * seq_stride + seq_coloff + cb] = hvv;
    if (seqH)
      *(v8h*)&seqH[((size_t)(b0 + r) * Tt + t) * seq_stride + seq_coloff + cb] = hhv;
    __syncthreads();
  }
  if (hfin)
    *(v8f*)&hfin[(size_t)(b0 + r) * h_stride + h_coloff + cb] =
        *(const v8f*)&hF[r][cb];
}

// ---------------- flash attention (32 q-rows / WG / head) ----------------
// Q,K row-major [b,t,h,kd] f16 ; V transposed [b,h,kd,t] f16.
__global__ __launch_bounds__(256) void attn_flash_kernel(
    const _Float16* __restrict__ Q, const _Float16* __restrict__ Kh,
    const _Float16* __restrict__ Vt, _Float16* __restrict__ O) {
  const int t0 = blockIdx.x * 32;
  const int h  = blockIdx.y;
  const int b  = blockIdx.z;
  __shared__ __align__(32) _Float16 Qf[2][4][32][16];  // Q   A-frag layout
  __shared__ __align__(32) _Float16 Kf[2][4][32][16];  // K^T B-frag layout
  __shared__ __align__(32) _Float16 Vf[8][32][16];     // V   B-frag layout
  __shared__ __align__(32) _Float16 Pf[2][32][16];     // P   A-frag layout
  __shared__ float Ss[32][33];
  __shared__ float pmax[32][8], psum[32][8];
  __shared__ float mrun[32], lrun[32], corr[32], rowmax[32], lpart[32];
  const int tid = threadIdx.x, wave = tid >> 5, l = tid & 31;
  const int qr = tid >> 3, qc = (tid & 7) << 4;  // 32 rows x 8 groups of 16
  {  // Q staging into A-frag layout
    const size_t base = (((size_t)b * Tt + t0 + qr) * NHh + h) * KDd + qc;
    const v8h g0 = *(const v8h*)&Q[base];
    const v8h g1 = *(const v8h*)&Q[base + 8];
    const int mtile = qr >> 4, m = qr & 15;
    const int a0 = qc, a1 = qc + 8;
    *(v8h*)&Qf[mtile][a0 >> 5][m + (((a0 >> 4) & 1) << 4)][a0 & 15] = g0;
    *(v8h*)&Qf[mtile][a1 >> 5][m + (((a1 >> 4) & 1) << 4)][a1 & 15] = g1;
  }
  if (tid < 32) { mrun[tid] = -1e30f; lrun[tid] = 0.0f; }
  v8f acc[2] = {};
  const int mt = wave & 1, ntA = (wave >> 1) << 1;
  const int n = l & 15, mh = l >> 4;
  const int srow = tid >> 3, sj = tid & 7, s0l = sj << 2;  // softmax mapping
  const _Float16* vtb = Vt + ((size_t)b * NHh + h) * KDd * Tt;
  const int vkd = tid >> 1, vsh = tid & 1;  // V staging mapping
  const float scale = 0.0883883476483184f;  // 1/sqrt(128)
  __syncthreads();

  for (int s0 = 0; s0 < Tt; s0 += 32) {
    {  // K^T staging: 16 contiguous kd -> one full B-frag record (async)
      const size_t kb = (((size_t)b * Tt + s0 + qr) * NHh + h) * KDd + qc;
      const int klane = (qr & 15) + (((qc >> 4) & 1) << 4);
      _Float16* rec = &Kf[qr >> 4][qc >> 5][klane][0];
      ASYNC_COPY_B128(&Kh[kb], rec);
      ASYNC_COPY_B128(&Kh[kb + 8], rec + 8);
    }
    {  // V staging from transposed V: 16 contiguous t -> one B-frag record
      const _Float16* src = &vtb[(size_t)vkd * Tt + s0 + (vsh << 4)];
      _Float16* rec = &Vf[vkd >> 4][(vkd & 15) + (vsh << 4)][0];
      ASYNC_COPY_B128(src, rec);
      ASYNC_COPY_B128(src + 8, rec + 8);
    }
    ASYNC_WAIT();
    __syncthreads();
    if (wave < 4) {  // S = Q K^T * scale : 2x2 tiles of 16x16
      const int smt = wave & 1, snt = wave >> 1;
      v8f sa = {};
#pragma unroll
      for (int ks = 0; ks < 4; ++ks) {
        v16h a  = *(const v16h*)&Qf[smt][ks][l][0];
        v16h bt = *(const v16h*)&Kf[snt][ks][l][0];
        sa = wmma_f16(a, bt, sa);
      }
#pragma unroll
      for (int i = 0; i < 8; ++i)
        Ss[smt * 16 + i + 8 * mh][snt * 16 + n] = sa[i] * scale;
    }
    __syncthreads();
    {  // softmax pass 1: per-row local max (8 threads/row)
      float m4 = Ss[srow][s0l];
      m4 = fmaxf(m4, Ss[srow][s0l + 1]);
      m4 = fmaxf(m4, Ss[srow][s0l + 2]);
      m4 = fmaxf(m4, Ss[srow][s0l + 3]);
      pmax[srow][sj] = m4;
    }
    __syncthreads();
    if (tid < 32) {  // reduce max, compute correction
      const float mo = mrun[tid];
      float mx = mo;
#pragma unroll
      for (int jj = 0; jj < 8; ++jj) mx = fmaxf(mx, pmax[tid][jj]);
      const float cf = __expf(mo - mx);
      mrun[tid] = mx;
      corr[tid] = cf;
      rowmax[tid] = mx;
      lpart[tid] = lrun[tid] * cf;
    }
    __syncthreads();
    {  // softmax pass 2: exp, write P in A-frag layout (v4h), partial sums
      const float mx = rowmax[srow];
      v4h pv;
      float sum = 0.0f;
#pragma unroll
      for (int q = 0; q < 4; ++q) {
        const float p = __expf(Ss[srow][s0l + q] - mx);
        pv[q] = (_Float16)p;
        sum += p;
      }
      const int plane = (srow & 15) + (((s0l >> 4) & 1) << 4);
      const int psb = s0l & 15;
      *(v4h*)&Pf[srow >> 4][plane][psb] = pv;
      psum[srow][sj] = sum;
    }
    __syncthreads();
    if (tid < 32) {  // finish row sums
      float li = lpart[tid];
#pragma unroll
      for (int jj = 0; jj < 8; ++jj) li += psum[tid][jj];
      lrun[tid] = li;
    }
    {  // rescale + acc += P @ V (all fragments vectorized)
      v16h a  = *(const v16h*)&Pf[mt][l][0];
      v16h b0 = *(const v16h*)&Vf[ntA][l][0];
      v16h b1 = *(const v16h*)&Vf[ntA + 1][l][0];
#pragma unroll
      for (int i = 0; i < 8; ++i) {
        const float cf = corr[mt * 16 + i + 8 * mh];
        acc[0][i] *= cf;
        acc[1][i] *= cf;
      }
      acc[0] = wmma_f16(a, b0, acc[0]);
      acc[1] = wmma_f16(a, b1, acc[1]);
    }
    __syncthreads();
  }
#pragma unroll
  for (int j = 0; j < 2; ++j) {
    const int nc = (ntA + j) * 16 + n;
#pragma unroll
    for (int i = 0; i < 8; ++i) {
      const int row = mt * 16 + i + 8 * mh;
      const float v = acc[j][i] / lrun[row];
      O[(((size_t)b * Tt + t0 + row) * NHh + h) * KDd + nc] = (_Float16)v;
    }
  }
}

// ---------------- dense head ----------------
__global__ __launch_bounds__(64) void dense_head_kernel(
    const float* __restrict__ hc, const float* __restrict__ w,
    const float* __restrict__ bsc, float* __restrict__ out) {
  const int b = blockIdx.x * blockDim.x + threadIdx.x;
  if (b >= Bsz) return;
  float acc = bsc[0];
  for (int c = 0; c < Mm; ++c) acc += hc[(size_t)b * Mm + c] * w[c];
  out[b] = acc;
}

// ---------------- host ----------------
extern "C" void kernel_launch(void* const* d_in, const int* in_sizes, int n_in,
                              void* d_out, int out_size, void* d_ws,
                              size_t ws_size, hipStream_t stream) {
  (void)in_sizes; (void)n_in; (void)out_size; (void)ws_size;
  const float* inputs  = (const float*)d_in[0];
  const float* conv_w  = (const float*)d_in[1];
  const float* conv_b  = (const float*)d_in[2];
  const float* bn_g    = (const float*)d_in[3];
  const float* bn_b    = (const float*)d_in[4];
  const float* bn_m    = (const float*)d_in[5];
  const float* bn_v    = (const float*)d_in[6];
  const float* g1f_wx  = (const float*)d_in[7];
  const float* g1f_wh  = (const float*)d_in[8];
  const float* g1f_b   = (const float*)d_in[9];
  const float* g1b_wx  = (const float*)d_in[10];
  const float* g1b_wh  = (const float*)d_in[11];
  const float* g1b_b   = (const float*)d_in[12];
  const float* wq      = (const float*)d_in[13];
  const float* bq      = (const float*)d_in[14];
  const float* wk      = (const float*)d_in[15];
  const float* bk      = (const float*)d_in[16];
  const float* wv      = (const float*)d_in[17];
  const float* bv      = (const float*)d_in[18];
  const float* wo      = (const float*)d_in[19];
  const float* bo      = (const float*)d_in[20];
  const float* g2f_wx  = (const float*)d_in[21];
  const float* g2f_wh  = (const float*)d_in[22];
  const float* g2f_b   = (const float*)d_in[23];
  const float* g2b_wx  = (const float*)d_in[24];
  const float* g2b_wh  = (const float*)d_in[25];
  const float* g2b_b   = (const float*)d_in[26];
  const float* dense_w = (const float*)d_in[27];
  const float* dense_b = (const float*)d_in[28];

  char* ws = (char*)d_ws;
  size_t off = 0;
  auto alloc = [&](size_t bytes) {
    size_t r = off;
    off = (off + bytes + 255) & ~(size_t)255;
    return r;
  };
  float*    xg1f  = (float*)(ws + alloc((size_t)BT * G3x * 4));  // also q / xg2f
  float*    xg1b  = (float*)(ws + alloc((size_t)BT * G3x * 4));  // also k / xg2b
  _Float16* vt    = (_Float16*)(ws + alloc((size_t)BT * QKVd * 2));  // V^T
  _Float16* oh    = (_Float16*)(ws + alloc((size_t)BT * QKVd * 2));
  _Float16* seq1h = (_Float16*)(ws + alloc((size_t)BT * Mm * 2));
  float*    seq1f = (float*)(ws + alloc((size_t)BT * Mm * 4));
  float*    x2f   = (float*)(ws + alloc((size_t)BT * Mm * 4));
  _Float16* x2h   = (_Float16*)(ws + alloc((size_t)BT * Mm * 2));
  _Float16* x1h   = (_Float16*)(ws + alloc((size_t)BT * Ff * 2));
  _Float16* inph  = (_Float16*)(ws + alloc((size_t)Bsz * (Tt + 2) * Dd * 2));
  float*    hcat  = (float*)(ws + alloc((size_t)Bsz * Mm * 4));
  float*    bnsc  = (float*)(ws + alloc((size_t)Ff * 4));
  float*    bnsh  = (float*)(ws + alloc((size_t)Ff * 4));
  _Float16* convw_p  = (_Float16*)(ws + alloc((size_t)3 * Dd * Ff * 2));
  _Float16* g1f_wx_p = (_Float16*)(ws + alloc((size_t)Ff * G3x * 2));
  _Float16* g1f_wh_p = (_Float16*)(ws + alloc((size_t)Uu * G3x * 2));
  _Float16* g1b_wx_p = (_Float16*)(ws + alloc((size_t)Ff * G3x * 2));
  _Float16* g1b_wh_p = (_Float16*)(ws + alloc((size_t)Uu * G3x * 2));
  _Float16* wq_p     = (_Float16*)(ws + alloc((size_t)Mm * QKVd * 2));
  _Float16* wk_p     = (_Float16*)(ws + alloc((size_t)Mm * QKVd * 2));
  _Float16* wv_p     = (_Float16*)(ws + alloc((size_t)Mm * QKVd * 2));
  _Float16* wo_p     = (_Float16*)(ws + alloc((size_t)QKVd * Mm * 2));
  _Float16* g2f_wx_p = (_Float16*)(ws + alloc((size_t)Mm * G3x * 2));
  _Float16* g2f_wh_p = (_Float16*)(ws + alloc((size_t)Uu * G3x * 2));
  _Float16* g2b_wx_p = (_Float16*)(ws + alloc((size_t)Mm * G3x * 2));
  _Float16* g2b_wh_p = (_Float16*)(ws + alloc((size_t)Uu * G3x * 2));
  _Float16* qh   = (_Float16*)xg1f;  // reuse after GRU1
  _Float16* kh   = (_Float16*)xg1b;
  float*    xg2f = xg1f;             // reuse after attention
  float*    xg2b = xg1b;

  auto pack = [&](const float* src, _Float16* dst, int K, int N) {
    pack_b_kernel<<<(K * N + 255) / 256, 256, 0, stream>>>(src, dst, K, N);
  };
  pack(conv_w, convw_p, 3 * Dd, Ff);  // conv as im2col GEMM: K=192, N=64
  pack(g1f_wx, g1f_wx_p, Ff, G3x);
  pack(g1f_wh, g1f_wh_p, Uu, G3x);
  pack(g1b_wx, g1b_wx_p, Ff, G3x);
  pack(g1b_wh, g1b_wh_p, Uu, G3x);
  pack(wq, wq_p, Mm, QKVd);
  pack(wk, wk_p, Mm, QKVd);
  pack(wv, wv_p, Mm, QKVd);
  pack(wo, wo_p, QKVd, Mm);
  pack(g2f_wx, g2f_wx_p, Mm, G3x);
  pack(g2f_wh, g2f_wh_p, Uu, G3x);
  pack(g2b_wx, g2b_wx_p, Mm, G3x);
  pack(g2b_wh, g2b_wh_p, Uu, G3x);

  prep_bn_kernel<<<1, Ff, 0, stream>>>(conv_b, bn_g, bn_b, bn_m, bn_v, bnsc, bnsh);
  cvt_pad_kernel<<<(Bsz * (Tt + 2) * Dd + 255) / 256, 256, 0, stream>>>(inputs,
                                                                        inph);

  // 1. conv (im2col WMMA GEMM, fused BN+ReLU) -> x1 (f16)
  gemm_wmma_kernel<2><<<dim3(1, BT / BMT), 256, 0, stream>>>(
      inph, convw_p, nullptr, nullptr, bnsc, bnsh, nullptr, x1h, nullptr, BT,
      3 * Dd, Ff, Dd, 2 * Dd);

  // 2. GRU1 input gates
  gemm_wmma_kernel<4><<<dim3(G3x / 128, BT / BMT), 256, 0, stream>>>(
      x1h, g1f_wx_p, g1f_b, nullptr, nullptr, nullptr, xg1f, nullptr, nullptr,
      BT, Ff, G3x, Ff, 0);
  gemm_wmma_kernel<4><<<dim3(G3x / 128, BT / BMT), 256, 0, stream>>>(
      x1h, g1b_wx_p, g1b_b, nullptr, nullptr, nullptr, xg1b, nullptr, nullptr,
      BT, Ff, G3x, Ff, 0);

  // 3. GRU1 recurrence -> seq1 (concat fwd|bwd), f32 + f16
  gru_seq_kernel<<<Bsz / 16, 256, 0, stream>>>(
      xg1f, g1f_wh_p, g1f_b + G3x, seq1f, seq1h, Mm, 0, nullptr, 0, 0, 0);
  gru_seq_kernel<<<Bsz / 16, 256, 0, stream>>>(
      xg1b, g1b_wh_p, g1b_b + G3x, seq1f, seq1h, Mm, Uu, nullptr, 0, 0, 1);

  // 4. QKV projections (V written transposed per head)
  gemm_wmma_kernel<4><<<dim3(QKVd / 128, BT / BMT), 256, 0, stream>>>(
      seq1h, wq_p, bq, nullptr, nullptr, nullptr, nullptr, qh, nullptr, BT, Mm,
      QKVd, Mm, 0);
  gemm_wmma_kernel<4><<<dim3(QKVd / 128, BT / BMT), 256, 0, stream>>>(
      seq1h, wk_p, bk, nullptr, nullptr, nullptr, nullptr, kh, nullptr, BT, Mm,
      QKVd, Mm, 0);
  gemm_wmma_kernel<4><<<dim3(QKVd / 128, BT / BMT), 256, 0, stream>>>(
      seq1h, wv_p, bv, nullptr, nullptr, nullptr, nullptr, nullptr, vt, BT, Mm,
      QKVd, Mm, 0);

  // 5. flash attention -> o (f16)
  attn_flash_kernel<<<dim3(Tt / 32, NHh, Bsz), 256, 0, stream>>>(qh, kh, vt, oh);

  // 6. output projection + residual -> x2 (f32 + f16)
  gemm_wmma_kernel<4><<<dim3(Mm / 128, BT / BMT), 256, 0, stream>>>(
      oh, wo_p, bo, seq1f, nullptr, nullptr, x2f, x2h, nullptr, BT, QKVd, Mm,
      QKVd, 0);

  // 7. GRU2 input gates
  gemm_wmma_kernel<4><<<dim3(G3x / 128, BT / BMT), 256, 0, stream>>>(
      x2h, g2f_wx_p, g2f_b, nullptr, nullptr, nullptr, xg2f, nullptr, nullptr,
      BT, Mm, G3x, Mm, 0);
  gemm_wmma_kernel<4><<<dim3(G3x / 128, BT / BMT), 256, 0, stream>>>(
      x2h, g2b_wx_p, g2b_b, nullptr, nullptr, nullptr, xg2b, nullptr, nullptr,
      BT, Mm, G3x, Mm, 0);

  // 8. GRU2 recurrence -> final states concat [B, 2U]
  gru_seq_kernel<<<Bsz / 16, 256, 0, stream>>>(
      xg2f, g2f_wh_p, g2f_b + G3x, nullptr, nullptr, 0, 0, hcat, Mm, 0, 0);
  gru_seq_kernel<<<Bsz / 16, 256, 0, stream>>>(
      xg2b, g2b_wh_p, g2b_b + G3x, nullptr, nullptr, 0, 0, hcat, Mm, Uu, 1);

  // 9. dense head -> [B,1]
  dense_head_kernel<<<1, 64, 0, stream>>>(hcat, dense_w, dense_b,
                                          (float*)d_out);
}
